// YOLOv3Head_82480551952556
// MI455X (gfx1250) — matrix operand order repaired and checked by reading, not compiled, not run on hardware
//
#include <hip/hip_runtime.h>
#include <hip/hip_bf16.h>

// ---------------------------------------------------------------------------
// MI455X (gfx1250) YOLOv3 head.
// Convs lowered to implicit GEMM on v_wmma_f32_16x16x32_bf16 (wave32).
// Head conv stages B tiles with GLOBAL_LOAD_ASYNC_TO_LDS (ASYNCcnt pipeline).
// Each wave owns TWO 16-cout tiles -> 8 WMMAs per 4 LDS B-fragments.
// Everything is L2-resident (<= 192MB); compute-bound on the matrix units.
// ---------------------------------------------------------------------------

typedef __bf16 bf16;
typedef __attribute__((ext_vector_type(16))) __bf16 v16bf;
typedef __attribute__((ext_vector_type(8)))  float  v8f;

__constant__ float c_anch[18] = {
    32.64f, 47.68f,  50.24f, 108.16f, 126.72f, 96.32f,
    78.4f,  201.92f, 178.24f,178.56f, 129.6f,  294.72f,
    331.84f,194.56f, 227.84f,325.76f, 365.44f, 358.72f};

// --------------------------- gfx1250 async helpers -------------------------
// CDNA5 ISA: GLOBAL_LOAD_ASYNC_TO_LDS_B128
//   VDST = LDS byte-offset VGPR, VADDR = 64-bit global address, tracked by
//   ASYNCcnt.  Inline asm (portable across toolchains per CDNA5_HIP.md).
__device__ __forceinline__ void async_g2l_b128(const void* g, void* l)
{
#if defined(__gfx1250__)
    asm volatile("global_load_async_to_lds_b128 %0, %1, off"
                 :: "v"((unsigned int)(size_t)l),          // LDS offset (low 32b of generic ptr)
                    "v"((unsigned long long)(size_t)g)     // global address (VGPR pair)
                 : "memory");
#else
    (void)g; (void)l;
#endif
}

#if defined(__gfx1250__)
#if __has_builtin(__builtin_amdgcn_s_wait_asynccnt)
#define WAIT_ASYNC(n) __builtin_amdgcn_s_wait_asynccnt(n)
#else
#define WAIT_ASYNC(n) asm volatile("s_wait_asynccnt %0" :: "i"(n))
#endif
#else
#define WAIT_ASYNC(n)
#endif

// ---------------------------------------------------------------------------
// Pack fp32 weights [COUT][CIN][TAPS] (tap-major K = tap*CIN + c) into the
// CDNA5 WMMA A-fragment lane layout (CDNA5 ISA):
//   lane L (M = L&15, half = L>>4), VGPR v holds K pair:
//   K = (v<4 ? 2v : 16 + 2(v-4)) + half*8 (+0/+1)
// Fragment f = (mtile*KC + kc): 32 lanes x 16 bf16 = 1KB, 32B per lane.
// ---------------------------------------------------------------------------
__global__ void pack_wmma_a(const float* __restrict__ Wsrc, bf16* __restrict__ dst,
                            int COUT, int CIN, int TAPS, int MPAD, int KC)
{
    int t = blockIdx.x * blockDim.x + threadIdx.x;
    int total = (MPAD / 16) * KC * 512;
    if (t >= total) return;
    int f    = t >> 9;
    int r    = t & 511;
    int lane = r >> 4;
    int j    = r & 15;
    int v    = j >> 1;
    int tt   = j & 1;
    int mt   = f / KC;
    int kc   = f - mt * KC;
    int M    = mt * 16 + (lane & 15);
    int half = lane >> 4;
    int klocal = ((v < 4) ? 0 : 16) + half * 8 + (v & 3) * 2 + tt;
    int K    = kc * 32 + klocal;
    int tap  = K / CIN;
    int c    = K - tap * CIN;
    float val = (M < COUT) ? Wsrc[((size_t)M * CIN + c) * TAPS + tap] : 0.f;
    dst[t] = (bf16)val;
}

// ---------------------------------------------------------------------------
// Fused 3x3 conv + BN + LeakyReLU as implicit GEMM.
// Block = 8 waves; tile = 256 couts (16 x M16, 2 per wave) x 64 pixels.
// Double-buffered LDS: gather chunk kc+1 while chunk kc's WMMAs execute.
// K-chunk = 32 channels of one tap; tap = kc >> log2(CIN/32) (shift, no div).
// ---------------------------------------------------------------------------
__global__ __launch_bounds__(256) void conv3x3_bn_lrelu_wmma(
    const float* __restrict__ x, const bf16* __restrict__ wpack,
    const float* __restrict__ bias, const float* __restrict__ gam,
    const float* __restrict__ bet,  const float* __restrict__ mu,
    const float* __restrict__ var,  bf16* __restrict__ y,
    int CIN, int COUT, int H, int W, int NPIX, int NPIXP, int KC, int ccsh)
{
    __shared__ __align__(32) bf16 Btile[2][32][80];   // 160B row stride, 32B subtiles

    const int tid  = threadIdx.x;
    const int lane = tid & 31;
    const int wave = tid >> 5;
    const int nBase = blockIdx.x * 64;
    const int coutBlock = blockIdx.y;
    const int b = blockIdx.z;
    const int wtile0 = coutBlock * 16 + wave;        // first 16-cout tile
    const int wtile1 = wtile0 + 8;                   // second 16-cout tile
    const int HW = H * W;
    const int cmask = (1 << ccsh) - 1;

    // Per-thread staging geometry: invariant across the whole K loop.
    const int srow = (tid * 8) >> 6;        // channel row within 32-chunk
    const int col0 = (tid * 8) & 63;        // first pixel column (8 consecutive)
    int  pys[8], pxs[8];
    bool nok[8];
    #pragma unroll
    for (int i = 0; i < 8; ++i) {
        int n = nBase + col0 + i;
        nok[i] = (n < NPIX);
        int py = n / W;
        pys[i] = py;
        pxs[i] = n - py * W;
    }
    const float* xb0 = x + (size_t)b * CIN * HW + (size_t)srow * HW;

    auto stage = [&](int kc, int buf) {
        int tap = kc >> ccsh;
        int c0  = (kc & cmask) << 5;
        int dyq = (tap * 11) >> 5;          // == tap/3 for tap in [0,9)
        int dy  = dyq - 1;
        int dx  = tap - dyq * 3 - 1;
        const float* xb = xb0 + (size_t)c0 * HW;
        __builtin_prefetch(xb + (size_t)32 * HW, 0, 3);   // speculative, next chunk
        #pragma unroll
        for (int i = 0; i < 8; ++i) {
            int sy = pys[i] + dy;
            int sx = pxs[i] + dx;
            float val = 0.f;
            if (nok[i] && (unsigned)sy < (unsigned)H && (unsigned)sx < (unsigned)W)
                val = xb[sy * W + sx];
            Btile[buf][srow][col0 + i] = (bf16)val;
        }
    };

    v8f acc0[4] = {};
    v8f acc1[4] = {};
    stage(0, 0);
    __syncthreads();

    for (int kc = 0; kc < KC; ++kc) {
        const int buf = kc & 1;
        // Overlap: gather next chunk into the other buffer while WMMAs run.
        if (kc + 1 < KC) stage(kc + 1, buf ^ 1);

        // Two A fragments (pre-swizzled, one 32B load per lane each).
        v16bf af0 = *(const v16bf*)(wpack + ((size_t)(wtile0 * KC + kc) * 32 + lane) * 16);
        v16bf af1 = *(const v16bf*)(wpack + ((size_t)(wtile1 * KC + kc) * 32 + lane) * 16);
        #pragma unroll
        for (int nt = 0; nt < 4; ++nt) {
            // B fragment: pixel N = lane&15, K in table order per element,
            // reused by both M tiles -> 2 WMMAs per fragment.
            v16bf bfr;
            #pragma unroll
            for (int s = 0; s < 16; ++s)
                bfr[s] = Btile[buf][((s & 8) << 1) + (lane >> 4) * 8 + (s & 7)][nt * 16 + (lane & 15)];
            acc0[nt] = __builtin_amdgcn_wmma_f32_16x16x32_bf16(
                false, af0, false, bfr, (short)0, acc0[nt], false, false);
            acc1[nt] = __builtin_amdgcn_wmma_f32_16x16x32_bf16(
                false, af1, false, bfr, (short)0, acc1[nt], false, false);
        }
        __syncthreads();   // writers of buf^1 done + readers of buf done
    }

    // Epilogue: C/D layout VGPR r -> M = r + 8*(lane>=16), N = lane&15.
    const int half = lane >> 4;
    const int ncol = lane & 15;
    #pragma unroll
    for (int part = 0; part < 2; ++part) {
        const v8f* acc = part ? acc1 : acc0;
        #pragma unroll
        for (int r = 0; r < 8; ++r) {
            int co = (coutBlock * 16 + wave + part * 8) * 16 + half * 8 + r;
            float scale = gam[co] * rsqrtf(var[co] + 1e-5f);
            float shift = (bias[co] - mu[co]) * scale + bet[co];
            #pragma unroll
            for (int nt = 0; nt < 4; ++nt) {
                int pix = nBase + nt * 16 + ncol;
                float t = acc[nt][r] * scale + shift;
                t = t > 0.f ? t : 0.1f * t;
                y[((size_t)b * COUT + co) * NPIXP + pix] = (bf16)t;
            }
        }
    }
}

// ---------------------------------------------------------------------------
// 1x1 head conv: WMMA GEMM, M = 255 padded to 256 (all 16 M-tiles in one
// block: 2 per wave).  B tile (32 ch x 64 pix, 4KB) staged once per block via
// async LDS loads: 256 lanes x global_load_async_to_lds_b128 = whole tile in
// one instruction, double-buffered on ASYNCcnt.
// ---------------------------------------------------------------------------
__global__ __launch_bounds__(256) void conv1x1_head_wmma(
    const bf16* __restrict__ y, const bf16* __restrict__ wpackH,
    const float* __restrict__ pb, float* __restrict__ pred,
    int C2, int NPIXP, int KC)
{
    __shared__ __align__(32) bf16 Yt[2][32][80];

    const int tid  = threadIdx.x;
    const int lane = tid & 31;
    const int wave = tid >> 5;
    const int nBase = blockIdx.x * 64;
    const int b = blockIdx.z;
    const int wtile0 = wave;        // M tiles 0..7
    const int wtile1 = wave + 8;    // M tiles 8..15

    const int srow = tid >> 3;         // channel row 0..31
    const int scol = (tid & 7) * 8;    // 8 halves = 16B segment
    const bf16* ybase = y + ((size_t)b * C2 + srow) * NPIXP + nBase + scol;

    v8f acc0[4] = {};
    v8f acc1[4] = {};

    // kick off chunk 0
    async_g2l_b128(ybase, &Yt[0][srow][scol]);

    for (int kc = 0; kc < KC; ++kc) {
        const int buf = kc & 1;
        if (kc + 1 < KC) {
            // stream next chunk into the other buffer (read-safe: barrier'd last iter)
            async_g2l_b128(ybase + (size_t)(kc + 1) * 32 * NPIXP,
                           &Yt[buf ^ 1][srow][scol]);
            WAIT_ASYNC(1);   // current chunk landed; next stays in flight
        } else {
            WAIT_ASYNC(0);
        }
        __syncthreads();

        v16bf af0 = *(const v16bf*)(wpackH + ((size_t)(wtile0 * KC + kc) * 32 + lane) * 16);
        v16bf af1 = *(const v16bf*)(wpackH + ((size_t)(wtile1 * KC + kc) * 32 + lane) * 16);
        #pragma unroll
        for (int nt = 0; nt < 4; ++nt) {
            v16bf bfr;
            #pragma unroll
            for (int s = 0; s < 16; ++s)
                bfr[s] = Yt[buf][((s & 8) << 1) + (lane >> 4) * 8 + (s & 7)][nt * 16 + (lane & 15)];
            acc0[nt] = __builtin_amdgcn_wmma_f32_16x16x32_bf16(
                false, af0, false, bfr, (short)0, acc0[nt], false, false);
            acc1[nt] = __builtin_amdgcn_wmma_f32_16x16x32_bf16(
                false, af1, false, bfr, (short)0, acc1[nt], false, false);
        }
        __syncthreads();   // readers of buf done before it is overwritten
    }

    const int half = lane >> 4;
    const int ncol = lane & 15;
    #pragma unroll
    for (int part = 0; part < 2; ++part) {
        const v8f* acc = part ? acc1 : acc0;
        #pragma unroll
        for (int r = 0; r < 8; ++r) {
            int m = (wave + part * 8) * 16 + half * 8 + r;
            if (m < 255) {
                float bv = pb[m];
                #pragma unroll
                for (int nt = 0; nt < 4; ++nt) {
                    int pix = nBase + nt * 16 + ncol;
                    pred[((size_t)b * 255 + m) * NPIXP + pix] = acc[nt][r] + bv;
                }
            }
        }
    }
}

// ---------------------------------------------------------------------------
// Box decode: sigmoid/softmax/exp, writes dets5 + cls + keep=0 and NMS scratch.
// ---------------------------------------------------------------------------
__global__ void decode_kernel(const float* __restrict__ pred1,
                              const float* __restrict__ pred2,
                              const float* __restrict__ pred3,
                              float* __restrict__ dets5, int* __restrict__ clsOut,
                              unsigned char* __restrict__ keepOut,
                              float* __restrict__ wBoxes, float* __restrict__ wSc,
                              int* __restrict__ wCls)
{
    const int NTOT = 25200;
    int g = blockIdx.x * blockDim.x + threadIdx.x;
    if (g >= 8 * NTOT) return;
    int b = g / NTOT;
    int n = g - b * NTOT;

    int lvl, base, W_, NPP, strd;
    const float* pred;
    if (n < 19200)      { lvl = 0; base = 0;     W_ = 80; NPP = 6400; strd = 8;  pred = pred1; }
    else if (n < 24000) { lvl = 1; base = 19200; W_ = 40; NPP = 1600; strd = 16; pred = pred2; }
    else                { lvl = 2; base = 24000; W_ = 20; NPP = 448;  strd = 32; pred = pred3; }

    int na  = n - base;
    int pix = na / 3;
    int a   = na - pix * 3;
    const float* pp = pred + (size_t)b * 255 * NPP + pix;   // channel stride = NPP

    float conf = pp[(size_t)a * NPP];

    // softmax max + argmax + sum-exp over 80 classes (two L2-resident passes)
    float mx = -1e30f; int am = 0;
    for (int c = 0; c < 80; ++c) {
        float l = pp[(size_t)(3 + a * 80 + c) * NPP];
        if (l > mx) { mx = l; am = c; }
    }
    float se = 0.f;
    for (int c = 0; c < 80; ++c)
        se += __expf(pp[(size_t)(3 + a * 80 + c) * NPP] - mx);
    float sig   = 1.f / (1.f + __expf(-conf));
    float score = sig / se;   // sigmoid(conf) * max(softmax)

    float tx = pp[(size_t)(243 + a * 4 + 0) * NPP];
    float ty = pp[(size_t)(243 + a * 4 + 1) * NPP];
    float tw = pp[(size_t)(243 + a * 4 + 2) * NPP];
    float th = pp[(size_t)(243 + a * 4 + 3) * NPP];

    int gx = pix % W_, gy = pix / W_;
    float cx = (1.f / (1.f + __expf(-tx)) + (float)gx) * (float)strd;
    float cy = (1.f / (1.f + __expf(-ty)) + (float)gy) * (float)strd;
    float bw = __expf(tw) * c_anch[(lvl * 3 + a) * 2 + 0];
    float bh = __expf(th) * c_anch[(lvl * 3 + a) * 2 + 1];

    float x1 = fminf(fmaxf((cx - bw * 0.5f) * (1.f / 640.f), 0.f), 1.f);
    float y1 = fminf(fmaxf((cy - bh * 0.5f) * (1.f / 640.f), 0.f), 1.f);
    float x2 = fminf(fmaxf((cx + bw * 0.5f) * (1.f / 640.f), 0.f), 1.f);
    float y2 = fminf(fmaxf((cy + bh * 0.5f) * (1.f / 640.f), 0.f), 1.f);

    size_t oi = (size_t)b * NTOT + n;
    dets5[oi * 5 + 0] = (x1 + x2) * 0.5f * 640.f;
    dets5[oi * 5 + 1] = (y1 + y2) * 0.5f * 640.f;
    dets5[oi * 5 + 2] = (x2 - x1) * 640.f;
    dets5[oi * 5 + 3] = (y2 - y1) * 640.f;
    dets5[oi * 5 + 4] = score;
    clsOut[oi] = am;
    keepOut[oi] = 0;

    wBoxes[oi * 4 + 0] = x1; wBoxes[oi * 4 + 1] = y1;
    wBoxes[oi * 4 + 2] = x2; wBoxes[oi * 4 + 3] = y2;
    wSc[oi]  = score;
    wCls[oi] = am;
}

// ---------------------------------------------------------------------------
// Per-image NMS: threshold filter -> bitonic sort (2048) -> greedy suppress.
// ---------------------------------------------------------------------------
__global__ __launch_bounds__(256) void nms_kernel(
    const float* __restrict__ boxes, const float* __restrict__ sc,
    const int* __restrict__ cls, unsigned char* __restrict__ keepOut)
{
    const int N = 25200, CAP = 2048, NTHR = 256;
    __shared__ float s_sc[CAP];
    __shared__ int   s_idx[CAP];
    __shared__ int   s_cnt;
    __shared__ float s_x1[1024], s_y1[1024], s_x2[1024], s_y2[1024], s_ar[1024];
    __shared__ unsigned char s_keep[1024];
    __shared__ int s_flag;

    int b = blockIdx.x, tid = threadIdx.x;
    if (tid == 0) s_cnt = 0;
    __syncthreads();

    for (int n = tid; n < N; n += NTHR) {
        float s = sc[(size_t)b * N + n];
        if (s >= 0.3f) {
            int p = atomicAdd(&s_cnt, 1);
            if (p < CAP) { s_sc[p] = s; s_idx[p] = n; }
        }
    }
    __syncthreads();
    int M = s_cnt < CAP ? s_cnt : CAP;
    for (int i = tid; i < CAP; i += NTHR)
        if (i >= M) { s_sc[i] = -1e30f; s_idx[i] = -1; }
    __syncthreads();

    // bitonic sort, descending by score
    for (int k = 2; k <= CAP; k <<= 1) {
        for (int j = k >> 1; j > 0; j >>= 1) {
            for (int t = tid; t < CAP; t += NTHR) {
                int ixj = t ^ j;
                if (ixj > t) {
                    bool desc = ((t & k) == 0);
                    float av = s_sc[t], bv = s_sc[ixj];
                    bool doSwap = desc ? (av < bv) : (av > bv);
                    if (doSwap) {
                        s_sc[t] = bv; s_sc[ixj] = av;
                        int ti = s_idx[t]; s_idx[t] = s_idx[ixj]; s_idx[ixj] = ti;
                    }
                }
            }
            __syncthreads();
        }
    }

    int M2 = M < 1000 ? M : 1000;
    for (int i = tid; i < M2; i += NTHR) {
        int id = s_idx[i];
        float off = (float)cls[(size_t)b * N + id] * 4.0f;
        float x1 = boxes[((size_t)b * N + id) * 4 + 0] + off;
        float y1 = boxes[((size_t)b * N + id) * 4 + 1] + off;
        float x2 = boxes[((size_t)b * N + id) * 4 + 2] + off;
        float y2 = boxes[((size_t)b * N + id) * 4 + 3] + off;
        s_x1[i] = x1; s_y1[i] = y1; s_x2[i] = x2; s_y2[i] = y2;
        s_ar[i] = (x2 - x1) * (y2 - y1);
        s_keep[i] = 0;
    }
    __syncthreads();

    for (int i = 0; i < M2; ++i) {
        if (tid == 0) s_flag = 0;
        __syncthreads();
        for (int j = tid; j < i; j += NTHR) {
            if (s_keep[j]) {
                float xx1 = fmaxf(s_x1[i], s_x1[j]);
                float yy1 = fmaxf(s_y1[i], s_y1[j]);
                float xx2 = fminf(s_x2[i], s_x2[j]);
                float yy2 = fminf(s_y2[i], s_y2[j]);
                float inter = fmaxf(1e-28f, xx2 - xx1) * fmaxf(1e-28f, yy2 - yy1);
                float iou = inter / (s_ar[i] + s_ar[j] - inter);
                if (iou > 0.5f) s_flag = 1;
            }
        }
        __syncthreads();
        if (tid == 0) s_keep[i] = (s_flag == 0) ? 1 : 0;
        __syncthreads();
    }

    for (int i = tid; i < M2; i += NTHR)
        if (s_keep[i]) keepOut[(size_t)b * N + s_idx[i]] = 1;
}

// ---------------------------------------------------------------------------
// Host orchestration
// ---------------------------------------------------------------------------
extern "C" void kernel_launch(void* const* d_in, const int* in_sizes, int n_in,
                              void* d_out, int out_size, void* d_ws, size_t ws_size,
                              hipStream_t stream)
{
    (void)in_sizes; (void)n_in; (void)out_size; (void)ws_size;

    const float* p[3] = {(const float*)d_in[0], (const float*)d_in[1], (const float*)d_in[2]};
    struct Lvl { const float *w, *b, *g, *be, *m, *v, *pw, *pb; } L[3];
    for (int i = 0; i < 3; ++i) {
        int k = 3 + i * 8;
        L[i].w  = (const float*)d_in[k + 0];
        L[i].b  = (const float*)d_in[k + 1];
        L[i].g  = (const float*)d_in[k + 2];
        L[i].be = (const float*)d_in[k + 3];
        L[i].m  = (const float*)d_in[k + 4];
        L[i].v  = (const float*)d_in[k + 5];
        L[i].pw = (const float*)d_in[k + 6];
        L[i].pb = (const float*)d_in[k + 7];
    }

    static const int CIN_[3]  = {128, 256, 512};
    static const int COUT_[3] = {256, 512, 1024};
    static const int HH_[3]   = {80, 40, 20};

    char* ws = (char*)d_ws;
    size_t off = 0;
    auto take = [&](size_t bytes) -> char* {
        char* ptr = ws + off;
        off = (off + bytes + 255) & ~(size_t)255;
        return ptr;
    };

    bf16*  wpack[3];  bf16* wpackH[3];  bf16* ybuf[3];  float* predb[3];
    int NPIX_[3], NPIXP_[3];
    for (int i = 0; i < 3; ++i) {
        int CIN = CIN_[i], COUT = COUT_[i], H = HH_[i];
        NPIX_[i]  = H * H;
        NPIXP_[i] = ((NPIX_[i] + 63) / 64) * 64;
        int KC  = 9 * CIN / 32;
        int KCh = COUT / 32;
        wpack[i]  = (bf16*)take((size_t)(COUT / 16) * KC * 512 * 2);
        wpackH[i] = (bf16*)take((size_t)16 * KCh * 512 * 2);
        ybuf[i]   = (bf16*)take((size_t)8 * COUT * NPIXP_[i] * 2);
        predb[i]  = (float*)take((size_t)8 * 255 * NPIXP_[i] * 4);
    }
    float* wBoxes = (float*)take((size_t)8 * 25200 * 4 * 4);
    float* wSc    = (float*)take((size_t)8 * 25200 * 4);
    int*   wCls   = (int*)  take((size_t)8 * 25200 * 4);

    for (int i = 0; i < 3; ++i) {
        int CIN = CIN_[i], COUT = COUT_[i], H = HH_[i];
        int NPIX = NPIX_[i], NPIXP = NPIXP_[i];
        int KC  = 9 * CIN / 32;
        int KCh = COUT / 32;
        int ccsh = __builtin_ctz(CIN / 32);   // 2, 3, 4

        int total = (COUT / 16) * KC * 512;
        pack_wmma_a<<<(total + 255) / 256, 256, 0, stream>>>(
            L[i].w, wpack[i], COUT, CIN, 9, COUT, KC);
        int totalH = 16 * KCh * 512;
        pack_wmma_a<<<(totalH + 255) / 256, 256, 0, stream>>>(
            L[i].pw, wpackH[i], 255, COUT, 1, 256, KCh);

        dim3 g1(NPIXP / 64, COUT / 256, 8);   // 256 couts per block (2 M-tiles/wave)
        conv3x3_bn_lrelu_wmma<<<g1, 256, 0, stream>>>(
            p[i], wpack[i], L[i].b, L[i].g, L[i].be, L[i].m, L[i].v,
            ybuf[i], CIN, COUT, H, H, NPIX, NPIXP, KC, ccsh);

        dim3 g2(NPIXP / 64, 1, 8);            // all 16 M-tiles in one block
        conv1x1_head_wmma<<<g2, 256, 0, stream>>>(
            ybuf[i], wpackH[i], L[i].pb, predb[i], COUT, NPIXP, KCh);
    }

    // d_out layout: dets5 (8*25200*5 f32) | cls_inds (8*25200 i32) | keep (8*25200 u8)
    float* dets5 = (float*)d_out;
    int*   clsOut = (int*)((float*)d_out + (size_t)8 * 25200 * 5);
    unsigned char* keepOut =
        (unsigned char*)((char*)d_out + ((size_t)8 * 25200 * 5 + (size_t)8 * 25200) * 4);

    int totDec = 8 * 25200;
    decode_kernel<<<(totDec + 255) / 256, 256, 0, stream>>>(
        predb[0], predb[1], predb[2], dets5, clsOut, keepOut, wBoxes, wSc, wCls);

    nms_kernel<<<8, 256, 0, stream>>>(wBoxes, wSc, wCls, keepOut);
}
